// SupervisedGraphsage_32804960207333
// MI455X (gfx1250) — matrix operand order repaired and checked by reading, not compiled, not run on hardware
//
#include <hip/hip_runtime.h>
#include <hip/hip_bf16.h>
#include <stdint.h>

// ---- ext-vector types (trivial, union-safe; avoid HIP_vector_type ctors) ----
typedef _Float16 half_t;
typedef __attribute__((ext_vector_type(16))) _Float16 v16h;
typedef __attribute__((ext_vector_type(8)))  float    v8f;
typedef __attribute__((ext_vector_type(4)))  float    f32x4;
typedef __attribute__((ext_vector_type(4)))  unsigned int u32x4;

#define NNODES  200000
#define MAXDEG  128
#define BATCH   1024
#define S1      25
#define S2      10
#define FDIM    256
#define HDIM    128
#define NCLS    41
#define M_HOP   (BATCH*S1)        // 25600 hop-1 rows
#define M1ROWS  (M_HOP + BATCH)   // 26624 total layer-1 rows (hop1 rows + self rows)

__device__ __forceinline__ uint32_t hash_u32(uint32_t x){
    x ^= x >> 16; x *= 0x7feb352du; x ^= x >> 15; x *= 0x846ca68bu; x ^= x >> 16; return x;
}
// deterministic hop-1 sample for flat row id m (b = m/S1)
__device__ __forceinline__ int samp1(const int* __restrict__ adj, int nid, int m){
    return adj[(size_t)nid*MAXDEG + (hash_u32(0x9E3779B9u ^ (uint32_t)m) & (MAXDEG-1))];
}

union H8  { half_t h[8]; u32x4 q; };
union AB  { v16h v; u32x4 q[2]; };

// ---------------------------------------------------------------------------
// Kernel 1: pack W matrices (f32 [256,128]) into wave32 WMMA B-fragment layout,
// fp16.  Index = ((nt*8 + kb)*32 + lane)*16 + j ; nt<8 = self, nt>=8 = neigh.
// B layout (16x16x32 f16): lanes 0-15 hold K = kb*32+{0..7,16..23} at col lane,
// lanes 16-31 hold K = kb*32+{8..15,24..31} at col lane-16.
// ---------------------------------------------------------------------------
__global__ void pack_w(const float* __restrict__ Ws1, const float* __restrict__ Wn1,
                       const float* __restrict__ Ws2, const float* __restrict__ Wn2,
                       half_t* __restrict__ WB1, half_t* __restrict__ WB2)
{
    int idx = blockIdx.x * blockDim.x + threadIdx.x;   // 0 .. 2*65536-1
    if (idx >= 2*65536) return;
    int layer = idx >> 16;
    int rem   = idx & 65535;
    int nt   = rem >> 12;          // 0..15
    int kb   = (rem >> 9) & 7;     // 0..7
    int lane = (rem >> 4) & 31;    // 0..31
    int j    = rem & 15;           // 0..15
    int n = (nt & 7)*16 + (lane & 15);
    int k = kb*32 + (lane >> 4)*8 + (j & 7) + (j >> 3)*16;
    const float* Wsrc = (layer == 0) ? (nt < 8 ? Ws1 : Wn1) : (nt < 8 ? Ws2 : Wn2);
    half_t* dst = (layer == 0) ? WB1 : WB2;
    dst[rem] = (half_t)Wsrc[k*HDIM + n];
}

// ---------------------------------------------------------------------------
// Kernel 2: gather + neighbor-mean staging (HBM-bound: ~290 MB of f32 reads).
// One wave per output row; lane owns 8 contiguous features.
// rows [0, M_HOP)        : XS = f1 (hop1 feature), XN = mean_{S2} f2
// rows [M_HOP, M1ROWS)   : XS = f0 (self feature), XN = mean_{S1} f1
// ---------------------------------------------------------------------------
__global__ void gather_stage(const int* __restrict__ node_ids,
                             const float* __restrict__ feat,
                             const int* __restrict__ adj,
                             half_t* __restrict__ XS, half_t* __restrict__ XN)
{
    int row  = (blockIdx.x * blockDim.x + threadIdx.x) >> 5;
    int lane = threadIdx.x & 31;
    if (row >= M1ROWS) return;
    int col = lane * 8;

    float acc[8];
#pragma unroll
    for (int t = 0; t < 8; ++t) acc[t] = 0.f;

    int selfNode;
    float scale;
    if (row < M_HOP) {
        int b = row / S1;
        selfNode = samp1(adj, node_ids[b], row);
        const int* arow = adj + (size_t)selfNode * MAXDEG;
        int nb = arow[hash_u32(0xB5297A4Du ^ (uint32_t)(row*S2)) & (MAXDEG-1)];
        for (int j = 0; j < S2; ++j) {
            const f32x4* np = (const f32x4*)(feat + (size_t)nb*FDIM + col);
            if (j + 1 < S2) {
                nb = arow[hash_u32(0xB5297A4Du ^ (uint32_t)(row*S2 + j + 1)) & (MAXDEG-1)];
                __builtin_prefetch(feat + (size_t)nb*FDIM + col, 0, 0); // global_prefetch_b8
            }
            f32x4 a = np[0], c = np[1];
#pragma unroll
            for (int t = 0; t < 4; ++t) { acc[t] += a[t]; acc[4+t] += c[t]; }
        }
        scale = 1.f / (float)S2;
    } else {
        int b = row - M_HOP;
        selfNode = node_ids[b];
        for (int i = 0; i < S1; ++i) {
            int nb = samp1(adj, selfNode, b*S1 + i);
            const f32x4* np = (const f32x4*)(feat + (size_t)nb*FDIM + col);
            f32x4 a = np[0], c = np[1];
#pragma unroll
            for (int t = 0; t < 4; ++t) { acc[t] += a[t]; acc[4+t] += c[t]; }
        }
        scale = 1.f / (float)S1;
    }

    // self features -> fp16
    const f32x4* sp = (const f32x4*)(feat + (size_t)selfNode*FDIM + col);
    f32x4 s0 = sp[0], s1 = sp[1];
    H8 os, on;
#pragma unroll
    for (int t = 0; t < 4; ++t) { os.h[t] = (half_t)s0[t]; os.h[4+t] = (half_t)s1[t]; }
#pragma unroll
    for (int t = 0; t < 8; ++t) on.h[t] = (half_t)(acc[t] * scale);
    *(u32x4*)(XS + (size_t)row*FDIM + col) = os.q;
    *(u32x4*)(XN + (size_t)row*FDIM + col) = on.q;
}

// ---------------------------------------------------------------------------
// Kernel 3: layer-1 GEMM, H1[m, 0:256] = relu([XS@Ws1 ; XN@Wn1]) via
// v_wmma_f32_16x16x32_f16.  Block = 4 waves, one 16-row M-tile.
// wave 0,1 -> self half (ntiles 0-7), wave 2,3 -> neigh half (ntiles 8-15).
// ---------------------------------------------------------------------------
__global__ void __launch_bounds__(128)
gemm_l1(const half_t* __restrict__ XS, const half_t* __restrict__ XN,
        const half_t* __restrict__ WB, half_t* __restrict__ Hout)
{
    __shared__ __align__(16) half_t As[16][256];
    __shared__ __align__(16) half_t An[16][256];
    int mt  = blockIdx.x;
    int tid = threadIdx.x;

    // cooperative stage: 16x256 fp16 tile = 512 u32x4 per matrix
    {
        const u32x4* gs = (const u32x4*)(XS + (size_t)mt*16*256);
        const u32x4* gn = (const u32x4*)(XN + (size_t)mt*16*256);
        u32x4* ls = (u32x4*)&As[0][0];
        u32x4* ln = (u32x4*)&An[0][0];
#pragma unroll
        for (int i = 0; i < 4; ++i) {
            ls[tid + i*128] = gs[tid + i*128];
            ln[tid + i*128] = gn[tid + i*128];
        }
    }
    __syncthreads();

    int wave = tid >> 5, lane = tid & 31;
    int sub  = lane & 15, hi = lane >> 4;
    const half_t (*A)[256] = (wave < 2) ? As : An;

    v8f zero = {0.f,0.f,0.f,0.f,0.f,0.f,0.f,0.f};
    v8f acc[4];
#pragma unroll
    for (int q = 0; q < 4; ++q) acc[q] = zero;

#pragma unroll
    for (int kb = 0; kb < 8; ++kb) {
        // A frag (ISA 16-bit A 16x32): lanes0-15 h0..7=K0..7,h8..15=K16..23;
        // lanes16-31 h0..7=K8..15, h8..15=K24..31
        AB a;
        a.q[0] = *(const u32x4*)&A[sub][kb*32 + hi*8];
        a.q[1] = *(const u32x4*)&A[sub][kb*32 + 16 + hi*8];
#pragma unroll
        for (int q = 0; q < 4; ++q) {
            int nt = wave*4 + q;
            const u32x4* bp = (const u32x4*)(WB + ((((nt*8 + kb)*32) + lane) << 4));
            AB b; b.q[0] = bp[0]; b.q[1] = bp[1];
            acc[q] = __builtin_amdgcn_wmma_f32_16x16x32_f16(
                false, a.v, false, b.v, (short)0, acc[q], false, false);
        }
    }

    // D layout: lanes0-15 vgpr r -> (M=r, N=lane); lanes16-31 -> (M=8+r, N=lane-16)
#pragma unroll
    for (int q = 0; q < 4; ++q) {
        int ncol = (wave*4 + q)*16 + sub;
#pragma unroll
        for (int r = 0; r < 8; ++r) {
            int m = mt*16 + hi*8 + r;
            float v = acc[q][r];
            v = v > 0.f ? v : 0.f;   // relu
            Hout[(size_t)m*256 + ncol] = (half_t)v;
        }
    }
}

// ---------------------------------------------------------------------------
// Kernel 4: XN2[b] = mean over S1 of h1 rows.  One wave per batch row.
// ---------------------------------------------------------------------------
__global__ void reduce_h1(const half_t* __restrict__ H1, half_t* __restrict__ XN2)
{
    int b    = (blockIdx.x * blockDim.x + threadIdx.x) >> 5;
    int lane = threadIdx.x & 31;
    if (b >= BATCH) return;
    int col = lane * 8;
    float acc[8];
#pragma unroll
    for (int t = 0; t < 8; ++t) acc[t] = 0.f;
    for (int i = 0; i < S1; ++i) {
        H8 u; u.q = *(const u32x4*)(H1 + (size_t)(b*S1 + i)*256 + col);
#pragma unroll
        for (int t = 0; t < 8; ++t) acc[t] += (float)u.h[t];
    }
    H8 o;
#pragma unroll
    for (int t = 0; t < 8; ++t) o.h[t] = (half_t)(acc[t] * (1.f/(float)S1));
    *(u32x4*)(XN2 + (size_t)b*256 + col) = o.q;
}

// ---------------------------------------------------------------------------
// Kernel 5: layer-2 GEMM (h0@Ws2 ; mean(h1)@Wn2) -> l2 normalize -> pred ->
// softmax.  Block = 4 waves per 16 batch rows.
// ---------------------------------------------------------------------------
__global__ void __launch_bounds__(128)
final_k(const half_t* __restrict__ H1, const half_t* __restrict__ XN2,
        const half_t* __restrict__ WB2, const float* __restrict__ Wp,
        const float* __restrict__ bpred, float* __restrict__ out)
{
    __shared__ __align__(16) half_t As[16][256];
    __shared__ __align__(16) half_t An[16][256];
    __shared__ float hbuf[16][256];
    __shared__ float red[16][8];
    __shared__ float scl[16];
    __shared__ float lg[16][48];

    int bt  = blockIdx.x;
    int tid = threadIdx.x;

    {   // self rows = H1[M_HOP + bt*16 ..], neigh rows = XN2[bt*16 ..]
        const u32x4* gs = (const u32x4*)(H1  + (size_t)(M_HOP + bt*16)*256);
        const u32x4* gn = (const u32x4*)(XN2 + (size_t)(bt*16)*256);
        u32x4* ls = (u32x4*)&As[0][0];
        u32x4* ln = (u32x4*)&An[0][0];
#pragma unroll
        for (int i = 0; i < 4; ++i) {
            ls[tid + i*128] = gs[tid + i*128];
            ln[tid + i*128] = gn[tid + i*128];
        }
    }
    __syncthreads();

    int wave = tid >> 5, lane = tid & 31;
    int sub  = lane & 15, hi = lane >> 4;
    const half_t (*A)[256] = (wave < 2) ? As : An;

    v8f zero = {0.f,0.f,0.f,0.f,0.f,0.f,0.f,0.f};
    v8f acc[4];
#pragma unroll
    for (int q = 0; q < 4; ++q) acc[q] = zero;

#pragma unroll
    for (int kb = 0; kb < 8; ++kb) {
        AB a;
        a.q[0] = *(const u32x4*)&A[sub][kb*32 + hi*8];
        a.q[1] = *(const u32x4*)&A[sub][kb*32 + 16 + hi*8];
#pragma unroll
        for (int q = 0; q < 4; ++q) {
            int nt = wave*4 + q;
            const u32x4* bp = (const u32x4*)(WB2 + ((((nt*8 + kb)*32) + lane) << 4));
            AB b; b.q[0] = bp[0]; b.q[1] = bp[1];
            acc[q] = __builtin_amdgcn_wmma_f32_16x16x32_f16(
                false, a.v, false, b.v, (short)0, acc[q], false, false);
        }
    }
#pragma unroll
    for (int q = 0; q < 4; ++q) {
        int ncol = (wave*4 + q)*16 + sub;
#pragma unroll
        for (int r = 0; r < 8; ++r)
            hbuf[hi*8 + r][ncol] = acc[q][r];   // identity activation
    }
    __syncthreads();

    // l2 normalize: 8 threads per row, 32 cols each
    int rr = tid >> 3, ss = tid & 7;
    {
        float p = 0.f;
        for (int c = ss*32; c < ss*32 + 32; ++c) p += hbuf[rr][c] * hbuf[rr][c];
        red[rr][ss] = p;
    }
    __syncthreads();
    if (ss == 0) {
        float t = 0.f;
        for (int k = 0; k < 8; ++k) t += red[rr][k];
        scl[rr] = rsqrtf(fmaxf(t, 1e-12f));
    }
    __syncthreads();
    for (int c = ss*32; c < ss*32 + 32; ++c) hbuf[rr][c] *= scl[rr];
    __syncthreads();

    // logits: 16*41 outputs, dot over 256
    for (int idx = tid; idx < 16*NCLS; idx += 128) {
        int r = idx / NCLS, c = idx % NCLS;
        float d = bpred[c];
        for (int k = 0; k < 256; ++k) d += hbuf[r][k] * Wp[k*NCLS + c];
        lg[r][c] = d;
    }
    __syncthreads();

    // softmax per row
    if (tid < 16) {
        int r = tid;
        float mx = -1e30f;
        for (int c = 0; c < NCLS; ++c) mx = fmaxf(mx, lg[r][c]);
        float sm = 0.f;
        for (int c = 0; c < NCLS; ++c) { float e = __expf(lg[r][c] - mx); lg[r][c] = e; sm += e; }
        float inv = 1.f / sm;
        for (int c = 0; c < NCLS; ++c)
            out[(size_t)(bt*16 + r)*NCLS + c] = lg[r][c] * inv;
    }
}

// ---------------------------------------------------------------------------
extern "C" void kernel_launch(void* const* d_in, const int* in_sizes, int n_in,
                              void* d_out, int out_size, void* d_ws, size_t ws_size,
                              hipStream_t stream)
{
    (void)in_sizes; (void)n_in; (void)out_size; (void)ws_size;
    const int*   node_ids = (const int*)  d_in[0];
    const float* features = (const float*)d_in[1];
    const int*   adj      = (const int*)  d_in[2];
    const float* Ws1      = (const float*)d_in[3];
    const float* Wn1      = (const float*)d_in[4];
    const float* Ws2      = (const float*)d_in[5];
    const float* Wn2      = (const float*)d_in[6];
    const float* Wp       = (const float*)d_in[7];
    const float* bp       = (const float*)d_in[8];
    float* out = (float*)d_out;

    char* ws = (char*)d_ws;
    size_t off = 0;
    auto alloc = [&](size_t bytes) -> void* {
        void* p = ws + off; off += (bytes + 255) & ~(size_t)255; return p;
    };
    half_t* XS1 = (half_t*)alloc((size_t)M1ROWS*256*sizeof(half_t)); // ~13.6 MB
    half_t* XN1 = (half_t*)alloc((size_t)M1ROWS*256*sizeof(half_t));
    half_t* H1  = (half_t*)alloc((size_t)M1ROWS*256*sizeof(half_t));
    half_t* XN2 = (half_t*)alloc((size_t)BATCH *256*sizeof(half_t));
    half_t* WB1 = (half_t*)alloc((size_t)65536*sizeof(half_t));
    half_t* WB2 = (half_t*)alloc((size_t)65536*sizeof(half_t));

    pack_w      <<<512, 256, 0, stream>>>(Ws1, Wn1, Ws2, Wn2, WB1, WB2);
    gather_stage<<<(M1ROWS*32 + 255)/256, 256, 0, stream>>>(node_ids, features, adj, XS1, XN1);
    gemm_l1     <<<M1ROWS/16, 128, 0, stream>>>(XS1, XN1, WB1, H1);
    reduce_h1   <<<(BATCH*32 + 255)/256, 256, 0, stream>>>(H1, XN2);
    final_k     <<<BATCH/16, 128, 0, stream>>>(H1, XN2, WB2, Wp, bp, out);
}
